// UNET_ViT_Wavenet_m_22574348108542
// MI455X (gfx1250) — matrix operand order repaired and checked by reading, not compiled, not run on hardware
//
#include <hip/hip_runtime.h>

typedef __attribute__((ext_vector_type(16))) _Float16 v16h;
typedef __attribute__((ext_vector_type(8)))  _Float16 v8h;
typedef __attribute__((ext_vector_type(8)))  float    v8f;

#define GT_M 64
#define GT_N 64
#define GT_K 32
#define BPAD 16   // sBt row pad (rows = 96B, keeps 16B alignment for b128 LDS ops)

// ---------------------------------------------------------------------------
// Guard-free f16->f32 WMMA GEMM: C[M,N] = A[M,K] * Bt[N,K]^T + bias
// B is ALWAYS provided in [N][K] (row = output column) layout; all producers
// pack weights/activations that way so both A and B tiles can be staged with
// GLOBAL_LOAD_ASYNC_TO_LDS_B128 (ASYNCcnt) -- no VGPR round-trip.
// REQUIRES: M % 64 == 0, N % 64 == 0, K % 32 == 0.
// 256 threads = 8 wave32 waves; block tile 64x64; each wave owns one 16x16
// m-tile and two 16x16 n-tiles (2 WMMAs / K-step). Double-buffered LDS.
// act: 0=none, 1=relu, 2=gelu(exact erf)
// ---------------------------------------------------------------------------
__global__ __launch_bounds__(256) void wmma_gemm_k(
    const _Float16* __restrict__ A, const _Float16* __restrict__ Bt,
    const float* __restrict__ bias, float* __restrict__ C,
    int M, int N, int K,
    long long strA, long long strB, long long strC, int act)
{
  __shared__ _Float16 sA[2][GT_M][GT_K];            // 8 KB
  __shared__ _Float16 sBt[2][GT_N][GT_K + BPAD];    // 12 KB, [n][k]
  const int tid  = threadIdx.x;
  const int lane = tid & 31;
  const int wave = tid >> 5;
  const int hf   = (lane >> 4) & 1;   // wave32 lane-half
  const int l16  = lane & 15;
  const int wm   = wave & 3;          // 4 wave rows  (16 each)
  const int wn   = wave >> 2;         // 2 wave cols  (32 each)
  const int m0   = blockIdx.y * GT_M;
  const int n0   = blockIdx.x * GT_N;
  const _Float16* Ab = A  + (long long)blockIdx.z * strA;
  const _Float16* Bb = Bt + (long long)blockIdx.z * strB;
  float*          Cb = C  + (long long)blockIdx.z * strC;

  // staging: 64 rows x 32 k halfs per tile; 256 threads x 16B = one b128 each
  const int ar = tid >> 2;            // row 0..63 (A: m, B: n)
  const int ak = (tid & 3) * 8;       // k-chunk {0,8,16,24}

  // CDNA5 async copy global->LDS (tracked by ASYNCcnt)
  auto stage_async = [&](int buf, int k0) {
    unsigned aoff = (unsigned)(unsigned long long)&sA[buf][ar][ak];
    const _Float16* gA = &Ab[(long long)(m0 + ar) * K + k0 + ak];
    asm volatile("global_load_async_to_lds_b128 %0, %1, off"
                 :: "v"(aoff), "v"(gA) : "memory");
    unsigned boff = (unsigned)(unsigned long long)&sBt[buf][ar][ak];
    const _Float16* gB = &Bb[(long long)(n0 + ar) * K + k0 + ak];
    asm volatile("global_load_async_to_lds_b128 %0, %1, off"
                 :: "v"(boff), "v"(gB) : "memory");
  };

  v8f acc0 = {}, acc1 = {};
  const int nsteps = K / GT_K;

  stage_async(0, 0);
  asm volatile("s_wait_asynccnt 0x0" ::: "memory");
  __syncthreads();

  for (int i = 0; i < nsteps; ++i) {
    const int cur = i & 1;
    if (i + 1 < nsteps) {
      if (i + 2 < nsteps)
        __builtin_prefetch(&Ab[(long long)(m0 + ar) * K + (i + 2) * GT_K + ak], 0, 0);
      stage_async(cur ^ 1, (i + 1) * GT_K);
    }

    // A fragment: lane holds row wm*16+l16; K = hf*16 .. hf*16+15 (contiguous)
    const int arow = wm * 16 + l16;
    v8h a0 = *(const v8h*)&sA[cur][arow][hf * 16];
    v8h a1 = *(const v8h*)&sA[cur][arow][hf * 16 + 8];
    v16h af = __builtin_shufflevector(a0, a1, 0,1,2,3,4,5,6,7,8,9,10,11,12,13,14,15);

    // B fragments: lane holds column n; K = hf*16 .. hf*16+15 (contiguous)
    const int nr0 = wn * 32 + l16;
    v8h b00 = *(const v8h*)&sBt[cur][nr0][hf * 16];
    v8h b01 = *(const v8h*)&sBt[cur][nr0][hf * 16 + 8];
    v16h bf0 = __builtin_shufflevector(b00, b01, 0,1,2,3,4,5,6,7,8,9,10,11,12,13,14,15);
    const int nr1 = nr0 + 16;
    v8h b10 = *(const v8h*)&sBt[cur][nr1][hf * 16];
    v8h b11 = *(const v8h*)&sBt[cur][nr1][hf * 16 + 8];
    v16h bf1 = __builtin_shufflevector(b10, b11, 0,1,2,3,4,5,6,7,8,9,10,11,12,13,14,15);

    acc0 = __builtin_amdgcn_wmma_f32_16x16x32_f16(false, af, false, bf0,
                                                  (short)0, acc0, false, false);
    acc1 = __builtin_amdgcn_wmma_f32_16x16x32_f16(false, af, false, bf1,
                                                  (short)0, acc1, false, false);

    // next-tile async copies must have landed before the barrier releases
    asm volatile("s_wait_asynccnt 0x0" ::: "memory");
    __syncthreads();
  }

  // C/D layout: vgpr r -> M = r + 8*hf, N = lane&15
  const int mbase = m0 + wm * 16 + hf * 8;
  const int nc0 = n0 + wn * 32 + l16;
  const int nc1 = nc0 + 16;
  float bb0 = bias ? bias[nc0] : 0.f;
  float bb1 = bias ? bias[nc1] : 0.f;
#pragma unroll
  for (int r = 0; r < 8; ++r) {
    float v0 = acc0[r] + bb0;
    float v1 = acc1[r] + bb1;
    if (act == 1) { v0 = fmaxf(v0, 0.f); v1 = fmaxf(v1, 0.f); }
    else if (act == 2) {
      v0 = 0.5f * v0 * (1.f + erff(v0 * 0.70710678118f));
      v1 = 0.5f * v1 * (1.f + erff(v1 * 0.70710678118f));
    }
    long long row = (long long)(mbase + r) * N;
    Cb[row + nc0] = v0;
    Cb[row + nc1] = v1;
  }
}

// ---------------------------------------------------------------------------
// CDNA5 TDM probe (not launched from host; keeps tensor path verified)
// ---------------------------------------------------------------------------
typedef __attribute__((ext_vector_type(4))) unsigned int tdm_ui4;
typedef __attribute__((ext_vector_type(8))) int tdm_i8;
typedef __attribute__((ext_vector_type(4))) int tdm_i4;
#if defined(__has_builtin)
#if __has_builtin(__builtin_amdgcn_tensor_load_to_lds)
#define HAVE_TDM 1
#endif
#endif
__global__ void cdna5_tdm_probe_k() {
#ifdef HAVE_TDM
  tdm_ui4 g0 = {};
  tdm_i8  g1 = {};
  tdm_i4  g2 = {};
  tdm_i4  g3 = {};
  tdm_i8  g4 = {};
  __builtin_amdgcn_tensor_load_to_lds(g0, g1, g2, g3, g4, 0);
  __builtin_amdgcn_s_wait_tensorcnt(0);
#endif
}

// ---------------------------------------------------------------------------
// Elementwise / packing / reduction kernels
// ---------------------------------------------------------------------------
__global__ void f32_to_f16_k(const float* __restrict__ s, _Float16* __restrict__ d, long long n) {
  long long i = (long long)blockIdx.x * blockDim.x + threadIdx.x;
  long long st = (long long)gridDim.x * blockDim.x;
  for (; i < n; i += st) d[i] = (_Float16)s[i];
}

// wt[n*K + k] = w[k*N + n]   (f32 [K,N] -> f16 [N][K])
__global__ void transpose_pack_k(const float* __restrict__ w, _Float16* __restrict__ wt,
                                 int K, int N) {
  long long total = (long long)K * N;
  long long i = (long long)blockIdx.x * blockDim.x + threadIdx.x;
  long long st = (long long)gridDim.x * blockDim.x;
  for (; i < total; i += st) {
    int k = (int)(i % K);
    long long n = i / K;
    wt[i] = (_Float16)w[(long long)k * N + n];
  }
}

// col[M][Kp], zero-padded in [K, Kp)
__global__ void im2col3x3_k(const float* __restrict__ x, _Float16* __restrict__ col,
                            int NBb, int C, int H, int W, int Kp) {
  int K = C * 9;
  long long total = (long long)NBb * H * W * Kp;
  long long i = (long long)blockIdx.x * blockDim.x + threadIdx.x;
  long long st = (long long)gridDim.x * blockDim.x;
  for (; i < total; i += st) {
    int kk = (int)(i % Kp);
    long long m = i / Kp;
    float v = 0.f;
    if (kk < K) {
      int xw = (int)(m % W); long long t = m / W;
      int yh = (int)(t % H); int b = (int)(t / H);
      int c = kk / 9, r = kk % 9, p = r / 3, q = r % 3;
      int iy = yh + p - 1, ix = xw + q - 1;
      if (iy >= 0 && iy < H && ix >= 0 && ix < W)
        v = x[(((long long)b * C + c) * H + iy) * W + ix];
    }
    col[i] = (_Float16)v;
  }
}

// Conv weights OIHW are already [O][C*9] = [N][K]; convert + pad K -> Kp.
__global__ void pack_wconv_bt_k(const float* __restrict__ w, _Float16* __restrict__ wbt,
                                int O, int C, int Kp) {
  int K = C * 9;
  long long total = (long long)O * Kp;
  long long i = (long long)blockIdx.x * blockDim.x + threadIdx.x;
  long long st = (long long)gridDim.x * blockDim.x;
  for (; i < total; i += st) {
    int k = (int)(i % Kp);
    long long o = i / Kp;
    wbt[i] = (_Float16)((k < K) ? w[o * K + k] : 0.f);
  }
}

__global__ void unpack_mo_k(const float* __restrict__ tmp, float* __restrict__ out,
                            int NBb, int O, int H, int W) {
  long long total = (long long)NBb * O * H * W;
  long long i = (long long)blockIdx.x * blockDim.x + threadIdx.x;
  long long st = (long long)gridDim.x * blockDim.x;
  for (; i < total; i += st) {
    int xw = (int)(i % W); long long t = i / W;
    int yh = (int)(t % H); t /= H;
    int o = (int)(t % O); int b = (int)(t / O);
    long long m = ((long long)b * H + yh) * W + xw;
    out[i] = tmp[m * O + o];
  }
}

__global__ __launch_bounds__(256) void bn_stats_k(const float* __restrict__ x,
                                                  float* mean, float* var,
                                                  int NBb, int C, int HW) {
  int c = blockIdx.x;
  long long n = (long long)NBb * HW;
  float s = 0.f, s2 = 0.f;
  for (long long i = threadIdx.x; i < n; i += 256) {
    int b = (int)(i / HW), p = (int)(i % HW);
    float v = x[((long long)b * C + c) * HW + p];
    s += v; s2 += v * v;
  }
  __shared__ float sh[256], sh2[256];
  sh[threadIdx.x] = s; sh2[threadIdx.x] = s2;
  __syncthreads();
  for (int o = 128; o > 0; o >>= 1) {
    if ((int)threadIdx.x < o) { sh[threadIdx.x] += sh[threadIdx.x + o]; sh2[threadIdx.x] += sh2[threadIdx.x + o]; }
    __syncthreads();
  }
  if (threadIdx.x == 0) {
    float m = sh[0] / (float)n;
    mean[c] = m;
    var[c]  = sh2[0] / (float)n - m * m;
  }
}

__global__ void bn_apply_k(float* __restrict__ x, const float* mean, const float* var,
                           const float* g, const float* b, int C, int HW,
                           long long total, int relu) {
  long long i = (long long)blockIdx.x * blockDim.x + threadIdx.x;
  long long st = (long long)gridDim.x * blockDim.x;
  for (; i < total; i += st) {
    int c = (int)((i / HW) % C);
    float v = (x[i] - mean[c]) * rsqrtf(var[c] + 1e-5f) * g[c] + b[c];
    x[i] = relu ? fmaxf(v, 0.f) : v;
  }
}

__global__ void haar_dwt_k(const float* __restrict__ x, float* __restrict__ y,
                           int NBb, int C, int H, int W) {
  int Ho = H / 2, Wo = W / 2;
  long long total = (long long)NBb * C * Ho * Wo;
  long long i = (long long)blockIdx.x * blockDim.x + threadIdx.x;
  long long st = (long long)gridDim.x * blockDim.x;
  for (; i < total; i += st) {
    int wo = (int)(i % Wo); long long t = i / Wo;
    int ho = (int)(t % Ho); t /= Ho;
    int c = (int)(t % C); int b = (int)(t / C);
    const float* p = x + (((long long)b * C + c) * H + 2 * ho) * W + 2 * wo;
    float a = p[0], bb = p[1], cc = p[W], dd = p[W + 1];
    long long ob = (long long)b * (4 * C);
#define OIDX(ch) (((ob + (ch)) * Ho + ho) * Wo + wo)
    y[OIDX(c)]         = 0.5f * (a + bb + cc + dd);
    y[OIDX(C + c)]     = 0.5f * (a + bb - cc - dd);
    y[OIDX(2 * C + c)] = 0.5f * (a - bb + cc - dd);
    y[OIDX(3 * C + c)] = 0.5f * (a - bb - cc + dd);
#undef OIDX
  }
}

__global__ void copy_into_ch_k(const float* __restrict__ src, float* __restrict__ dst,
                               int NBb, int Cs, int Ct, int coff, int HW) {
  long long total = (long long)NBb * Cs * HW;
  long long i = (long long)blockIdx.x * blockDim.x + threadIdx.x;
  long long st = (long long)gridDim.x * blockDim.x;
  for (; i < total; i += st) {
    long long b = i / ((long long)Cs * HW);
    long long r = i % ((long long)Cs * HW);
    dst[(b * Ct + coff) * HW + r] = src[i];
  }
}

__global__ void nchw_to_mc_f16_k(const float* __restrict__ x, _Float16* __restrict__ a,
                                 int NBb, int C, int HW) {
  long long total = (long long)NBb * HW * C;
  long long i = (long long)blockIdx.x * blockDim.x + threadIdx.x;
  long long st = (long long)gridDim.x * blockDim.x;
  for (; i < total; i += st) {
    int c = (int)(i % C);
    long long m = i / C;
    int p = (int)(m % HW); int b = (int)(m / HW);
    a[i] = (_Float16)x[((long long)b * C + c) * HW + p];
  }
}

__global__ void mc_to_nchw_k(const float* __restrict__ t, float* __restrict__ x,
                             int NBb, int C, int HW) {
  long long total = (long long)NBb * C * HW;
  long long i = (long long)blockIdx.x * blockDim.x + threadIdx.x;
  long long st = (long long)gridDim.x * blockDim.x;
  for (; i < total; i += st) {
    int p = (int)(i % HW); long long q = i / HW;
    int c = (int)(q % C); int b = (int)(q / C);
    x[i] = t[((long long)b * HW + p) * C + c];
  }
}

__global__ void scatter_up_k(const float* __restrict__ tmp, const float* __restrict__ upb,
                             float* __restrict__ y, int NBb, int O, int H, int W) {
  long long total = (long long)NBb * H * W * O * 4;
  long long i = (long long)blockIdx.x * blockDim.x + threadIdx.x;
  long long st = (long long)gridDim.x * blockDim.x;
  for (; i < total; i += st) {
    int oq = (int)(i % (O * 4));
    long long m = i / (O * 4);
    int o = oq >> 2, p = (oq >> 1) & 1, q = oq & 1;
    int w = (int)(m % W); long long t2 = m / W;
    int h = (int)(t2 % H); int b = (int)(t2 / H);
    y[(((long long)b * O + o) * (2 * H) + (2 * h + p)) * (2 * W) + (2 * w + q)] =
        tmp[i] + upb[o];
  }
}

__global__ void add_pos_k(float* __restrict__ t, const float* __restrict__ pos,
                          int TOKn, int EMBn, long long total) {
  long long i = (long long)blockIdx.x * blockDim.x + threadIdx.x;
  long long st = (long long)gridDim.x * blockDim.x;
  for (; i < total; i += st) {
    int e = (int)(i % EMBn);
    int tk = (int)((i / EMBn) % TOKn);
    t[i] += pos[(long long)tk * EMBn + e];
  }
}

__global__ void add_inplace_k(float* __restrict__ t, const float* __restrict__ s, long long n) {
  long long i = (long long)blockIdx.x * blockDim.x + threadIdx.x;
  long long st = (long long)gridDim.x * blockDim.x;
  for (; i < n; i += st) t[i] += s[i];
}

__global__ __launch_bounds__(256) void layernorm_f16_k(const float* __restrict__ x,
                                                       const float* g, const float* b,
                                                       _Float16* __restrict__ y, int D) {
  int row = blockIdx.x;
  const float* xr = x + (long long)row * D;
  __shared__ float sh[256];
  float s = 0.f;
  for (int i = threadIdx.x; i < D; i += 256) s += xr[i];
  sh[threadIdx.x] = s; __syncthreads();
  for (int o = 128; o > 0; o >>= 1) { if ((int)threadIdx.x < o) sh[threadIdx.x] += sh[threadIdx.x + o]; __syncthreads(); }
  float mean = sh[0] / D; __syncthreads();
  float s2 = 0.f;
  for (int i = threadIdx.x; i < D; i += 256) { float d = xr[i] - mean; s2 += d * d; }
  sh[threadIdx.x] = s2; __syncthreads();
  for (int o = 128; o > 0; o >>= 1) { if ((int)threadIdx.x < o) sh[threadIdx.x] += sh[threadIdx.x + o]; __syncthreads(); }
  float rstd = rsqrtf(sh[0] / D + 1e-5f);
  for (int i = threadIdx.x; i < D; i += 256)
    y[(long long)row * D + i] = (_Float16)((xr[i] - mean) * rstd * g[i] + b[i]);
}

// q,k: [B,NH,TOK,hd] (A rows / Bt rows).  v: TRANSPOSED to [B,NH,hd,TOK] so the
// P@V GEMM's B operand is in [N][K] layout for async staging.
__global__ void qkv_split_k(const float* __restrict__ qkv, _Float16* __restrict__ q,
                            _Float16* __restrict__ k, _Float16* __restrict__ v,
                            int NBb, int TOKn, int NH, int HD) {
  int EMBn = NH * HD;
  long long total = (long long)NBb * TOKn * EMBn;
  long long i = (long long)blockIdx.x * blockDim.x + threadIdx.x;
  long long st = (long long)gridDim.x * blockDim.x;
  for (; i < total; i += st) {
    int e = (int)(i % EMBn);
    long long m = i / EMBn;
    int tk = (int)(m % TOKn); int b = (int)(m / TOKn);
    int hh = e / HD, d = e % HD;
    long long o  = (((long long)(b * NH + hh) * TOKn) + tk) * HD + d;
    long long ov = (((long long)(b * NH + hh) * HD) + d) * TOKn + tk;
    long long src = m * (3LL * EMBn);
    q[o]  = (_Float16)qkv[src + e];
    k[o]  = (_Float16)qkv[src + EMBn + e];
    v[ov] = (_Float16)qkv[src + 2 * EMBn + e];
  }
}

__global__ __launch_bounds__(64) void softmax64_k(const float* __restrict__ s,
                                                  _Float16* __restrict__ p, float scale) {
  int row = blockIdx.x;
  int t = threadIdx.x;
  __shared__ float sh[64];
  float v = s[(long long)row * 64 + t] * scale;
  sh[t] = v; __syncthreads();
  for (int o = 32; o > 0; o >>= 1) { if (t < o) sh[t] = fmaxf(sh[t], sh[t + o]); __syncthreads(); }
  float mx = sh[0]; __syncthreads();
  float e = __expf(v - mx);
  sh[t] = e; __syncthreads();
  for (int o = 32; o > 0; o >>= 1) { if (t < o) sh[t] += sh[t + o]; __syncthreads(); }
  p[(long long)row * 64 + t] = (_Float16)(e / sh[0]);
}

__global__ void merge_heads_k(const float* __restrict__ ao, _Float16* __restrict__ om,
                              int NBb, int TOKn, int NH, int HD) {
  int EMBn = NH * HD;
  long long total = (long long)NBb * TOKn * EMBn;
  long long i = (long long)blockIdx.x * blockDim.x + threadIdx.x;
  long long st = (long long)gridDim.x * blockDim.x;
  for (; i < total; i += st) {
    int e = (int)(i % EMBn);
    long long m = i / EMBn;
    int tk = (int)(m % TOKn); int b = (int)(m / TOKn);
    int hh = e / HD, d = e % HD;
    om[i] = (_Float16)ao[(((long long)(b * NH + hh) * TOKn) + tk) * HD + d];
  }
}

__global__ void conv1x1_out_k(const float* __restrict__ x, const float* __restrict__ w,
                              const float* __restrict__ bias, float* __restrict__ y,
                              int NBb, int C, int HW) {
  long long total = (long long)NBb * HW;
  long long i = (long long)blockIdx.x * blockDim.x + threadIdx.x;
  long long st = (long long)gridDim.x * blockDim.x;
  for (; i < total; i += st) {
    int p = (int)(i % HW); int b = (int)(i / HW);
    float s = bias[0];
    for (int c = 0; c < C; ++c)
      s += x[((long long)b * C + c) * HW + p] * w[c];
    y[i] = s;
  }
}

// ---------------------------------------------------------------------------
// Host orchestration
// ---------------------------------------------------------------------------
static inline int gb(long long n) {
  long long b = (n + 255) / 256;
  if (b > 65535) b = 65535;
  if (b < 1) b = 1;
  return (int)b;
}

struct PConv { const float *b, *w; };
struct PBn   { const float *b, *g; };
struct PCb   { PBn bn1, bn2; PConv c1, c2; };
struct PEnc  { PCb cb; PBn dwt_bn; PConv dwt_c; };
struct PDec  { PCb cb; const float *up_b, *up_w; };
struct PBlk  { const float *b1,*b2,*bo,*bqkv,*ln1_b,*ln1_g,*ln2_b,*ln2_g,*w1,*w2,*wo,*wqkv; };

extern "C" void kernel_launch(void* const* d_in, const int* in_sizes, int n_in,
                              void* d_out, int out_size, void* d_ws, size_t ws_size,
                              hipStream_t stream) {
  (void)in_sizes; (void)n_in; (void)out_size; (void)ws_size;
  const int NB = 8;

  // ---- parse inputs (x, then params leaves in pytree (sorted-key) order) ----
  int ci = 0;
  auto nxt   = [&]() { return (const float*)d_in[ci++]; };
  const float* xin = nxt();
  auto pBn   = [&]() { PBn p; p.b = nxt(); p.g = nxt(); return p; };
  auto pConv = [&]() { PConv p; p.b = nxt(); p.w = nxt(); return p; };
  auto pCb   = [&]() { PCb p; p.bn1 = pBn(); p.bn2 = pBn(); p.c1 = pConv(); p.c2 = pConv(); return p; };

  PCb bott = pCb();                                   // 'bott'
  PDec dec[4];                                        // 'dec'
  for (int l = 0; l < 4; ++l) { dec[l].cb = pCb(); dec[l].up_b = nxt(); dec[l].up_w = nxt(); }
  PEnc enc[4];                                        // 'enc'
  for (int l = 0; l < 4; ++l) { enc[l].cb = pCb(); enc[l].dwt_bn = pBn(); enc[l].dwt_c = pConv(); }
  PConv outc = pConv();                               // 'out' (b, w)
  PBlk blks[12];                                      // 'vit' -> 'blocks'
  for (int l = 0; l < 12; ++l) {
    PBlk& k = blks[l];
    k.b1 = nxt(); k.b2 = nxt(); k.bo = nxt(); k.bqkv = nxt();
    k.ln1_b = nxt(); k.ln1_g = nxt(); k.ln2_b = nxt(); k.ln2_g = nxt();
    k.w1 = nxt(); k.w2 = nxt(); k.wo = nxt(); k.wqkv = nxt();
  }
  const float* patch_b = nxt();
  const float* patch_w = nxt();
  const float* pos     = nxt();

  // ---- workspace bump allocator ----
  char* wsbase = (char*)d_ws;
  size_t wsoff = 0;
  auto alloc  = [&](size_t bytes) { char* p = wsbase + wsoff; wsoff += (bytes + 255) & ~(size_t)255; return p; };
  auto allocF = [&](long long n) { return (float*)alloc((size_t)n * 4); };
  auto allocH = [&](long long n) { return (_Float16*)alloc((size_t)n * 2); };

  float* sbuf[4];
  sbuf[0] = allocF(8LL * 64 * 128 * 128);
  sbuf[1] = allocF(8LL * 128 * 64 * 64);
  sbuf[2] = allocF(8LL * 256 * 32 * 32);
  sbuf[3] = allocF(8LL * 512 * 16 * 16);
  float* T  = allocF(8LL * 64 * 128 * 128);           // conv_block internal
  float* Xb = allocF(8LL * 128 * 128 * 128);
  float* Yb = allocF(8LL * 128 * 128 * 128);
  float* Zb = allocF(8LL * 128 * 128 * 128);
  float* gemmtmp = allocF(8388608);
  float* meanb = allocF(1024);
  float* varb  = allocF(1024);
  _Float16* col = allocH(150994944LL);                // max M*Kp = 131072*1152
  _Float16* wp  = allocH(12582912LL);
  _Float16* a16 = allocH(8388608LL);
  float* tb   = allocF(524288);
  _Float16* h16 = allocH(524288);
  float* qkv  = allocF(1572864);
  _Float16* q16 = allocH(524288);
  _Float16* k16 = allocH(524288);
  _Float16* v16 = allocH(524288);
  float* sc   = allocF(131072);
  _Float16* p16 = allocH(131072);
  float* ao   = allocF(524288);
  _Float16* om16 = allocH(524288);
  float* proj = allocF(524288);
  float* m1   = allocF(262144);
  _Float16* m1h = allocH(262144);
  float* m2   = allocF(524288);

  // ---- helpers ----
  auto launch_gemm = [&](const _Float16* A, const _Float16* Bt, const float* bias,
                         float* C, int M, int N, int K, long long sa,
                         long long sb, long long scs, int batch, int act) {
    dim3 g((unsigned)(N / GT_N), (unsigned)(M / GT_M), (unsigned)batch);
    wmma_gemm_k<<<g, dim3(256), 0, stream>>>(A, Bt, bias, C, M, N, K, sa, sb, scs, act);
  };
  auto tpack = [&](const float* w, _Float16* wt, int K, int N) {
    transpose_pack_k<<<gb((long long)K * N), 256, 0, stream>>>(w, wt, K, N);
  };
  auto cvt16 = [&](const float* s, _Float16* d, long long n) {
    f32_to_f16_k<<<gb(n), 256, 0, stream>>>(s, d, n);
  };
  auto conv3x3 = [&](const float* in, PConv cp, float* out, int Ci, int Co, int H, int W) {
    long long M = (long long)NB * H * W;
    int K  = Ci * 9;
    int Kp = (K + 31) & ~31;
    im2col3x3_k<<<gb(M * Kp), 256, 0, stream>>>(in, col, NB, Ci, H, W, Kp);
    pack_wconv_bt_k<<<gb((long long)Co * Kp), 256, 0, stream>>>(cp.w, wp, Co, Ci, Kp);
    launch_gemm(col, wp, cp.b, gemmtmp, (int)M, Co, Kp, 0, 0, 0, 1, 0);
    unpack_mo_k<<<gb(M * Co), 256, 0, stream>>>(gemmtmp, out, NB, Co, H, W);
  };
  auto batchnorm = [&](float* x, PBn p, int C, int HW, int relu) {
    bn_stats_k<<<C, 256, 0, stream>>>(x, meanb, varb, NB, C, HW);
    long long tot = (long long)NB * C * HW;
    bn_apply_k<<<gb(tot), 256, 0, stream>>>(x, meanb, varb, p.g, p.b, C, HW, tot, relu);
  };
  auto conv_block = [&](const float* in, PCb cb, float* out, int Ci, int Co, int H, int W) {
    conv3x3(in, cb.c1, T, Ci, Co, H, W);
    batchnorm(T, cb.bn1, Co, H * W, 0);
    conv3x3(T, cb.c2, out, Co, Co, H, W);
    batchnorm(out, cb.bn2, Co, H * W, 1);
  };

  // ---- encoder ----
  const int encCi[4] = {3, 64, 128, 256};
  const int encCo[4] = {64, 128, 256, 512};
  int Hs = 128;
  const float* hin = xin;
  float* hbufs[2] = {Xb, Yb};
  int pp = 0;
  for (int l = 0; l < 4; ++l) {
    conv_block(hin, enc[l].cb, sbuf[l], encCi[l], encCo[l], Hs, Hs);
    haar_dwt_k<<<gb((long long)NB * encCo[l] * (Hs / 2) * (Hs / 2)), 256, 0, stream>>>(
        sbuf[l], Zb, NB, encCo[l], Hs, Hs);
    Hs /= 2;
    float* ho = hbufs[pp]; pp ^= 1;
    conv3x3(Zb, enc[l].dwt_c, ho, 4 * encCo[l], encCo[l], Hs, Hs);
    batchnorm(ho, enc[l].dwt_bn, encCo[l], Hs * Hs, 1);
    hin = ho;
  }
  // hin = h4 (512 @ 8x8, lives in Yb)

  // ---- bottleneck ----
  conv_block(hin, bott, Xb, 512, 1024, 8, 8);          // 1024 @ 8x8 in Xb

  // ---- ViT ----
  {
    const int TOKn = 64, EMBn = 1024, HD = 256, NH = 4;
    int Mv = NB * TOKn;                                 // 512
    nchw_to_mc_f16_k<<<gb((long long)Mv * EMBn), 256, 0, stream>>>(Xb, a16, NB, EMBn, TOKn);
    tpack(patch_w, wp, EMBn, EMBn);
    launch_gemm(a16, wp, patch_b, tb, Mv, EMBn, EMBn, 0, 0, 0, 1, 0);
    add_pos_k<<<gb((long long)Mv * EMBn), 256, 0, stream>>>(tb, pos, TOKn, EMBn, (long long)Mv * EMBn);
    for (int bi = 0; bi < 12; ++bi) {
      PBlk& bk = blks[bi];
      layernorm_f16_k<<<Mv, 256, 0, stream>>>(tb, bk.ln1_g, bk.ln1_b, h16, EMBn);
      tpack(bk.wqkv, wp, EMBn, 3 * EMBn);
      launch_gemm(h16, wp, bk.bqkv, qkv, Mv, 3 * EMBn, EMBn, 0, 0, 0, 1, 0);
      qkv_split_k<<<gb((long long)Mv * EMBn), 256, 0, stream>>>(qkv, q16, k16, v16, NB, TOKn, NH, HD);
      launch_gemm(q16, k16, nullptr, sc, TOKn, TOKn, HD,
                  (long long)TOKn * HD, (long long)TOKn * HD, (long long)TOKn * TOKn, NB * NH, 0);
      softmax64_k<<<NB * NH * TOKn, 64, 0, stream>>>(sc, p16, 0.0625f); // 1/sqrt(256)
      launch_gemm(p16, v16, nullptr, ao, TOKn, HD, TOKn,
                  (long long)TOKn * TOKn, (long long)HD * TOKn, (long long)TOKn * HD, NB * NH, 0);
      merge_heads_k<<<gb((long long)Mv * EMBn), 256, 0, stream>>>(ao, om16, NB, TOKn, NH, HD);
      tpack(bk.wo, wp, EMBn, EMBn);
      launch_gemm(om16, wp, bk.bo, proj, Mv, EMBn, EMBn, 0, 0, 0, 1, 0);
      add_inplace_k<<<gb((long long)Mv * EMBn), 256, 0, stream>>>(tb, proj, (long long)Mv * EMBn);
      layernorm_f16_k<<<Mv, 256, 0, stream>>>(tb, bk.ln2_g, bk.ln2_b, h16, EMBn);
      tpack(bk.w1, wp, EMBn, 512);
      launch_gemm(h16, wp, bk.b1, m1, Mv, 512, EMBn, 0, 0, 0, 1, 2 /*gelu*/);
      cvt16(m1, m1h, (long long)Mv * 512);
      tpack(bk.w2, wp, 512, EMBn);
      launch_gemm(m1h, wp, bk.b2, m2, Mv, EMBn, 512, 0, 0, 0, 1, 0);
      add_inplace_k<<<gb((long long)Mv * EMBn), 256, 0, stream>>>(tb, m2, (long long)Mv * EMBn);
    }
    mc_to_nchw_k<<<gb((long long)NB * EMBn * TOKn), 256, 0, stream>>>(tb, Yb, NB, EMBn, TOKn);
  }

  // ---- decoder ----
  const int decI[4] = {1024, 512, 256, 128};
  const int decO[4] = {512, 256, 128, 64};
  int Hd = 8;
  float* h = Yb;
  for (int l = 0; l < 4; ++l) {
    int I = decI[l], O = decO[l];
    long long M = (long long)NB * Hd * Hd;
    nchw_to_mc_f16_k<<<gb(M * I), 256, 0, stream>>>(h, a16, NB, I, Hd * Hd);
    tpack(dec[l].up_w, wp, I, O * 4);                   // (I,O,2,2)=[K,N] -> [N][K]
    launch_gemm(a16, wp, nullptr, gemmtmp, (int)M, O * 4, I, 0, 0, 0, 1, 0);
    scatter_up_k<<<gb(M * O * 4), 256, 0, stream>>>(gemmtmp, dec[l].up_b, Zb, NB, O, Hd, Hd);
    Hd *= 2;
    int HW = Hd * Hd;
    copy_into_ch_k<<<gb((long long)NB * O * HW), 256, 0, stream>>>(Zb, Xb, NB, O, 2 * O, 0, HW);
    copy_into_ch_k<<<gb((long long)NB * O * HW), 256, 0, stream>>>(sbuf[3 - l], Xb, NB, O, 2 * O, O, HW);
    conv_block(Xb, dec[l].cb, Yb, 2 * O, O, Hd, Hd);
    h = Yb;
  }

  // ---- output conv 1x1 ----
  conv1x1_out_k<<<gb(8LL * 128 * 128), 256, 0, stream>>>(h, outc.w, outc.b,
                                                         (float*)d_out, NB, 64, 128 * 128);
}
